// SetTransformerHSP_7198365188343
// MI455X (gfx1250) — matrix operand order repaired and checked by reading, not compiled, not run on hardware
//
#include <hip/hip_runtime.h>
#include <hip/hip_bf16.h>

typedef __attribute__((ext_vector_type(16))) __bf16 v16bf;
typedef __attribute__((ext_vector_type(8)))  __bf16 v8bf;
typedef __attribute__((ext_vector_type(2)))  __bf16 v2bf;
typedef __attribute__((ext_vector_type(8)))  float  v8f;
typedef __attribute__((ext_vector_type(4)))  float  v4f;

static __device__ __forceinline__ __bf16 f2bf(float f) {
  unsigned u = __builtin_bit_cast(unsigned, f);
  unsigned short h = (unsigned short)((u + 0x7FFFu + ((u >> 16) & 1u)) >> 16);
  return __builtin_bit_cast(__bf16, h);
}

// Packed f32x2 -> bf16x2 (RNE). Uses V_CVT_PK_BF16_F32 when the toolchain
// declares it; manual round-to-nearest-even otherwise (also host pass).
static __device__ __forceinline__ v2bf pk2(float lo, float hi) {
#if __has_builtin(__builtin_amdgcn_cvt_pk_bf16_f32)
  return __builtin_amdgcn_cvt_pk_bf16_f32(lo, hi);
#else
  v2bf r; r[0] = f2bf(lo); r[1] = f2bf(hi); return r;
#endif
}

#define BF16_WMMA(A_, B_, C_) \
  __builtin_amdgcn_wmma_f32_16x16x32_bf16(false, (A_), false, (B_), (short)0, (C_), false, false)

// ---------------------------------------------------------------------------
// Pack a row-major f32 (K x N) weight into the WMMA B-fragment-native bf16
// layout: dst[((kt*tilesN + nt)*32 + lane)*16 + e] = B[kt*32+(e>>3)*16+(lane>>4)*8+(e&7)][nt*16+(lane&15)]
// K zero-padded to a multiple of 32.
// ---------------------------------------------------------------------------
__global__ void pack_b_kernel(const float* __restrict__ src, __bf16* __restrict__ dst,
                              int K, int N) {
  const int tiles_n = N >> 4;
  const int Kt = (K + 31) >> 5;
  const long total = (long)Kt * tiles_n * 512;
  for (long idx = (long)blockIdx.x * blockDim.x + threadIdx.x; idx < total;
       idx += (long)gridDim.x * blockDim.x) {
    int e = (int)(idx & 15);
    int lane = (int)((idx >> 4) & 31);
    long rest = idx >> 9;
    int nt = (int)(rest % tiles_n);
    int kt = (int)(rest / tiles_n);
    int k = kt * 32 + (e >> 3) * 16 + (lane >> 4) * 8 + (e & 7);
    int n = nt * 16 + (lane & 15);
    float v = (k < K) ? src[(long)k * N + n] : 0.0f;
    dst[idx] = f2bf(v);
  }
}

// Pad x (1024 x 48) -> (1024 x 64) with zero columns so GEMM K is 32-aligned.
__global__ void pad_x_kernel(const float* __restrict__ x, float* __restrict__ xp) {
  int t = blockIdx.x * blockDim.x + threadIdx.x;   // 1024*64 total
  if (t < 1024 * 64) {
    int row = t >> 6, c = t & 63;
    xp[t] = (c < 48) ? x[row * 48 + c] : 0.0f;
  }
}

// Build a 16x32 bf16 A-fragment from 4 float4 registers.
static __device__ __forceinline__ v16bf make_a(v4f a0, v4f a1, v4f a2, v4f a3) {
  v16bf a;
#pragma unroll
  for (int e = 0; e < 2; ++e) {
    v2bf p0 = pk2(a0[2 * e], a0[2 * e + 1]);
    v2bf p1 = pk2(a1[2 * e], a1[2 * e + 1]);
    v2bf p2 = pk2(a2[2 * e], a2[2 * e + 1]);
    v2bf p3 = pk2(a3[2 * e], a3[2 * e + 1]);
    a[2 * e] = p0[0];      a[2 * e + 1] = p0[1];
    a[4 + 2 * e] = p1[0];  a[5 + 2 * e] = p1[1];
    a[8 + 2 * e] = p2[0];  a[9 + 2 * e] = p2[1];
    a[12 + 2 * e] = p3[0]; a[13 + 2 * e] = p3[1];
  }
  return a;
}

// ---------------------------------------------------------------------------
// Generic WMMA GEMM: C = epi(A(M x KT*32, f32, lda = K) @ Bpacked + bias)
// epi 0: none; 1: relu; 2: addsrc + relu(.)
// One 16x16 output tile per wave, 8 waves / block. unroll 2 pipelines loads
// across two WMMAs without blowing register pressure.
// ---------------------------------------------------------------------------
template <int KT>
__global__ void gemm_wmma_kernel(const float* __restrict__ A, const __bf16* __restrict__ Bp,
                                 const float* __restrict__ bias, const float* __restrict__ addsrc,
                                 float* __restrict__ C, int M, int N, int epi) {
  const int wave = threadIdx.x >> 5;
  const int lane = threadIdx.x & 31;
  const int kh = lane >> 4;
  const int ln = lane & 15;
  const int tiles_n = N >> 4;
  const int tiles = (M >> 4) * tiles_n;
  const int tile = blockIdx.x * (blockDim.x >> 5) + wave;
  if (tile >= tiles) return;               // uniform per wave: EXEC stays all-ones
  const int mt = tile / tiles_n;
  const int nt = tile - mt * tiles_n;
  const float* Arow = A + (long)(mt * 16 + ln) * (KT * 32) + kh * 8;
  v8f acc = {};
#pragma unroll 2
  for (int kt = 0; kt < KT; ++kt) {
    const float* Ak = Arow + kt * 32;
    v4f a0 = *reinterpret_cast<const v4f*>(Ak);
    v4f a1 = *reinterpret_cast<const v4f*>(Ak + 4);
    v4f a2 = *reinterpret_cast<const v4f*>(Ak + 16);
    v4f a3 = *reinterpret_cast<const v4f*>(Ak + 20);
    v16bf b = *reinterpret_cast<const v16bf*>(
        Bp + ((long)(kt * tiles_n + nt) * 32 + lane) * 16);
    acc = BF16_WMMA(make_a(a0, a1, a2, a3), b, acc);
  }
  const int n = nt * 16 + ln;
  const float bv = bias ? bias[n] : 0.0f;
#pragma unroll
  for (int v = 0; v < 8; ++v) {
    int row = mt * 16 + kh * 8 + v;
    float val = acc[v] + bv;
    if (epi >= 1) val = fmaxf(val, 0.0f);
    if (epi == 2) val += addsrc[(long)row * N + n];
    C[(long)row * N + n] = val;
  }
}

// ---------------------------------------------------------------------------
// Fused edge-MLP + adjacency aggregation:
//   agg[b,i,:] = sum_j adj[b,i,j] * relu( relu(hWi[b,i,:]+hWj[b,j,:]) @ W2 + b2 )
// (geb1 folded into hWi). One block (512 thr = 16 waves) per (b,i); each wave
// owns one 16-column tile so its 8 W2 fragments (64 VGPRs) stay register-
// resident across all 16 j-chunks -> no spills. 16-row j chunks staged as
// bf16 in LDS (stride 264 -> conflict-free fragment reads).
// ---------------------------------------------------------------------------
__global__ void edge_agg_kernel(const float* __restrict__ hWi, const float* __restrict__ hWj,
                                const float* __restrict__ adj, const __bf16* __restrict__ W2p,
                                const float* __restrict__ b2, float* __restrict__ agg) {
  const int bb = blockIdx.x >> 8;
  const int i  = blockIdx.x & 255;
  const int t = threadIdx.x;          // 0..511
  const int wave = t >> 5;            // 0..15 = column tile index
  const int lane = t & 31;
  const int kh = lane >> 4;
  const int ln = lane & 15;

  __shared__ __bf16 P[16][264];
  __shared__ float hi[256];
  __shared__ float arow[256];

  if (t < 256) {
    hi[t]   = hWi[((long)(bb * 256 + i)) * 256 + t];
    arow[t] = adj[((long)bb * 256 + i) * 256 + t];
  }

  // Hoist this wave's W2 fragments: 8 x v16bf = 64 VGPRs, loop-invariant.
  v16bf bw[8];
#pragma unroll
  for (int kt = 0; kt < 8; ++kt)
    bw[kt] = *reinterpret_cast<const v16bf*>(
        W2p + ((long)(kt * 16 + wave) * 32 + lane) * 16);

  __syncthreads();

  const int nc = wave * 16 + ln;
  const float bias = b2[nc];
  float part = 0.0f;

  for (int jc = 0; jc < 16; ++jc) {
    __syncthreads();  // previous chunk's fragment reads done before P rewrite
    {
      const int r  = t >> 5;          // 0..15 (row)
      const int c0 = lane << 3;       // 0..248, 8 cols per thread
      const float* hj = hWj + ((long)(bb * 256) + jc * 16 + r) * 256 + c0;
      v4f h0 = *reinterpret_cast<const v4f*>(hj);
      v4f h1 = *reinterpret_cast<const v4f*>(hj + 4);
      v2bf q0 = pk2(fmaxf(hi[c0 + 0] + h0[0], 0.f), fmaxf(hi[c0 + 1] + h0[1], 0.f));
      v2bf q1 = pk2(fmaxf(hi[c0 + 2] + h0[2], 0.f), fmaxf(hi[c0 + 3] + h0[3], 0.f));
      v2bf q2 = pk2(fmaxf(hi[c0 + 4] + h1[0], 0.f), fmaxf(hi[c0 + 5] + h1[1], 0.f));
      v2bf q3 = pk2(fmaxf(hi[c0 + 6] + h1[2], 0.f), fmaxf(hi[c0 + 7] + h1[3], 0.f));
      v8bf pr;
      pr[0] = q0[0]; pr[1] = q0[1]; pr[2] = q1[0]; pr[3] = q1[1];
      pr[4] = q2[0]; pr[5] = q2[1]; pr[6] = q3[0]; pr[7] = q3[1];
      *reinterpret_cast<v8bf*>(&P[r][c0]) = pr;   // one ds_store_b128
    }
    __syncthreads();

    v8f acc = {};
#pragma unroll
    for (int kt = 0; kt < 8; ++kt) {
      v8bf lo = *reinterpret_cast<const v8bf*>(&P[ln][kt * 32 + kh * 8]);
      v8bf hv = *reinterpret_cast<const v8bf*>(&P[ln][kt * 32 + 16 + kh * 8]);
      v16bf a;
#pragma unroll
      for (int e = 0; e < 8; ++e) { a[e] = lo[e]; a[8 + e] = hv[e]; }
      acc = BF16_WMMA(a, bw[kt], acc);
    }
#pragma unroll
    for (int v = 0; v < 8; ++v) {
      const float aw = arow[jc * 16 + kh * 8 + v];
      part += aw * fmaxf(acc[v] + bias, 0.0f);
    }
  }
  part += __shfl_xor(part, 16, 32);
  if (lane < 16) agg[(long)(bb * 256 + i) * 256 + nc] = part;
}

// ---------------------------------------------------------------------------
// Attention core: O = q + softmax(q k^T / 16) v  (8 heads, dh=32)
// One block per (b, head, qi).
// ---------------------------------------------------------------------------
__global__ void attention_kernel(const float* __restrict__ q, const float* __restrict__ k,
                                 const float* __restrict__ v, float* __restrict__ O,
                                 int Lq, int Lk) {
  const int qi = blockIdx.x % Lq;
  const int h  = (blockIdx.x / Lq) & 7;
  const int b  = blockIdx.x / (Lq * 8);
  const int t  = threadIdx.x;

  __shared__ float sc[256];
  __shared__ float red[256];
  __shared__ float qv[32];

  const float* qrow = q + ((long)(b * Lq + qi)) * 256 + h * 32;
  if (t < 32) qv[t] = qrow[t];
  __syncthreads();

  for (int kj = t; kj < Lk; kj += blockDim.x) {
    const float* krow = k + ((long)(b * Lk + kj)) * 256 + h * 32;
    float s = 0.0f;
#pragma unroll
    for (int d = 0; d < 32; ++d) s += qv[d] * krow[d];
    sc[kj] = s * 0.0625f;  // 1/sqrt(256)
  }
  __syncthreads();

  float m = -3.0e38f;
  for (int kj = t; kj < Lk; kj += blockDim.x) m = fmaxf(m, sc[kj]);
  red[t] = m;
  __syncthreads();
  for (int s = 128; s > 0; s >>= 1) {
    if (t < s) red[t] = fmaxf(red[t], red[t + s]);
    __syncthreads();
  }
  const float mx = red[0];
  __syncthreads();

  float sum = 0.0f;
  for (int kj = t; kj < Lk; kj += blockDim.x) {
    float e = __expf(sc[kj] - mx);
    sc[kj] = e;
    sum += e;
  }
  red[t] = sum;
  __syncthreads();
  for (int s = 128; s > 0; s >>= 1) {
    if (t < s) red[t] += red[t + s];
    __syncthreads();
  }
  const float inv = 1.0f / red[0];
  __syncthreads();

  if (t < 32) {
    float a = 0.0f;
    for (int kj = 0; kj < Lk; ++kj)
      a += sc[kj] * v[((long)(b * Lk + kj)) * 256 + h * 32 + t];
    O[((long)(b * Lq + qi)) * 256 + h * 32 + t] = qv[t] + a * inv;
  }
}

// Broadcast S (1,16,256) -> (4,16,256)
__global__ void bcast_S_kernel(const float* __restrict__ S, float* __restrict__ Qb) {
  int t = blockIdx.x * blockDim.x + threadIdx.x;
  if (t < 4 * 4096) Qb[t] = S[t & 4095];
}

// ---------------------------------------------------------------------------
extern "C" void kernel_launch(void* const* d_in, const int* in_sizes, int n_in,
                              void* d_out, int out_size, void* d_ws, size_t ws_size,
                              hipStream_t stream) {
  (void)in_sizes; (void)n_in; (void)out_size; (void)ws_size;
  const float* x     = (const float*)d_in[0];
  const float* adj   = (const float*)d_in[1];
  const float* gin_W = (const float*)d_in[2];
  const float* gin_b = (const float*)d_in[3];
  const float* geW1  = (const float*)d_in[4];
  const float* geb1  = (const float*)d_in[5];
  const float* geW2  = (const float*)d_in[6];
  const float* geb2  = (const float*)d_in[7];
  const float* gnW1  = (const float*)d_in[8];
  const float* gnb1  = (const float*)d_in[9];
  const float* gnW2  = (const float*)d_in[10];
  const float* gnb2  = (const float*)d_in[11];
  const float* goW   = (const float*)d_in[12];
  const float* gob   = (const float*)d_in[13];
  const float* S     = (const float*)d_in[14];
  const float* mWq   = (const float*)d_in[15];
  const float* mbq   = (const float*)d_in[16];
  const float* mWk   = (const float*)d_in[17];
  const float* mbk   = (const float*)d_in[18];
  const float* mWv   = (const float*)d_in[19];
  const float* mbv   = (const float*)d_in[20];
  const float* mWo   = (const float*)d_in[21];
  const float* mbo   = (const float*)d_in[22];
  const float* finW  = (const float*)d_in[23];
  const float* finb  = (const float*)d_in[24];

  char* wsp = (char*)d_ws;
  auto alloc_f = [&](size_t elems) -> float* {
    float* p = (float*)wsp; wsp += ((elems * 4 + 255) / 256) * 256; return p; };
  auto alloc_b = [&](size_t elems) -> __bf16* {
    __bf16* p = (__bf16*)wsp; wsp += ((elems * 2 + 255) / 256) * 256; return p; };
  auto pack_elems = [](int K, int N) -> size_t {
    return (size_t)((K + 31) / 32) * (N / 16) * 512; };

  __bf16* p_ginW = alloc_b(pack_elems(48, 256));
  __bf16* p_Wi   = alloc_b(65536);
  __bf16* p_Wj   = alloc_b(65536);
  __bf16* p_geW2 = alloc_b(65536);
  __bf16* p_gnW1 = alloc_b(65536);
  __bf16* p_gnW2 = alloc_b(65536);
  __bf16* p_goW  = alloc_b(65536);
  __bf16* p_m[12];
  for (int i = 0; i < 12; ++i) p_m[i] = alloc_b(65536);
  __bf16* p_finW = alloc_b(pack_elems(256, 384));

  const size_t RH = 1024 * 256;
  float* xpad = alloc_f(1024 * 64);
  float* big0 = alloc_f(RH);
  float* big1 = alloc_f(RH);
  float* big2 = alloc_f(RH);
  float* big3 = alloc_f(RH);
  float* big4 = alloc_f(RH);
  float* Qb = alloc_f(16384);
  float* qb = alloc_f(16384);
  float* Ob = alloc_f(16384);
  float* o1 = alloc_f(16384);
  float* o2 = alloc_f(16384);
  float* o3 = alloc_f(16384);

  auto pack = [&](const float* src, __bf16* dst, int K, int N) {
    long total = (long)pack_elems(K, N);
    int blocks = (int)((total + 255) / 256);
    pack_b_kernel<<<blocks, 256, 0, stream>>>(src, dst, K, N);
  };
  auto gemm = [&](const float* A, const __bf16* Bp, const float* bias, const float* add,
                  float* C, int M, int N, int K, int epi) {
    int tiles = (M / 16) * (N / 16);
    int blocks = (tiles + 7) / 8;
    if (K == 64)
      gemm_wmma_kernel<2><<<blocks, 256, 0, stream>>>(A, Bp, bias, add, C, M, N, epi);
    else
      gemm_wmma_kernel<8><<<blocks, 256, 0, stream>>>(A, Bp, bias, add, C, M, N, epi);
  };

  // ---- pack all weights into WMMA-fragment layout (bf16) ----
  pack(gin_W, p_ginW, 48, 256);
  pack(geW1, p_Wi, 256, 256);
  pack(geW1 + 256 * 256, p_Wj, 256, 256);
  pack(geW2, p_geW2, 256, 256);
  pack(gnW1, p_gnW1, 256, 256);
  pack(gnW2, p_gnW2, 256, 256);
  pack(goW, p_goW, 256, 256);
  for (int l = 0; l < 3; ++l) {
    pack(mWq + (long)l * 65536, p_m[l * 4 + 0], 256, 256);
    pack(mWk + (long)l * 65536, p_m[l * 4 + 1], 256, 256);
    pack(mWv + (long)l * 65536, p_m[l * 4 + 2], 256, 256);
    pack(mWo + (long)l * 65536, p_m[l * 4 + 3], 256, 256);
  }
  pack(finW, p_finW, 256, 384);

  // ---- trunk ----
  pad_x_kernel<<<256, 256, 0, stream>>>(x, xpad);
  float* h = big0;                                 // h = x @ gin_W + b
  gemm(xpad, p_ginW, gin_b, nullptr, h, 1024, 256, 64, 0);
  float* hWi = big1;                               // + geb1 folded in
  gemm(h, p_Wi, geb1, nullptr, hWi, 1024, 256, 256, 0);
  float* hWj = big2;
  gemm(h, p_Wj, nullptr, nullptr, hWj, 1024, 256, 256, 0);
  float* agg = big3;
  edge_agg_kernel<<<1024, 512, 0, stream>>>(hWi, hWj, adj, p_geW2, geb2, agg);
  float* t1 = big4;
  gemm(agg, p_gnW1, gnb1, nullptr, t1, 1024, 256, 256, 1);
  float* node = big0;                              // h dead
  gemm(t1, p_gnW2, gnb2, nullptr, node, 1024, 256, 256, 1);
  float* enc = big3;                               // agg dead
  gemm(node, p_goW, gob, nullptr, enc, 1024, 256, 256, 0);

  bcast_S_kernel<<<64, 256, 0, stream>>>(S, Qb);

  float* kb = big1;                                // hWi dead
  float* vb = big2;                                // hWj dead

  // ---- MAB 1: Q = S broadcast, K = enc (Lk = 256) ----
  gemm(Qb, p_m[0], mbq, nullptr, qb, 64, 256, 256, 0);
  gemm(enc, p_m[1], mbk, nullptr, kb, 1024, 256, 256, 0);
  gemm(enc, p_m[2], mbv, nullptr, vb, 1024, 256, 256, 0);
  attention_kernel<<<512, 256, 0, stream>>>(qb, kb, vb, Ob, 16, 256);
  gemm(Ob, p_m[3], mbo, Ob, o1, 64, 256, 256, 2);

  // ---- MAB 2: self (Lq = Lk = 16) ----
  gemm(o1, p_m[4], mbq + 256, nullptr, qb, 64, 256, 256, 0);
  gemm(o1, p_m[5], mbk + 256, nullptr, kb, 64, 256, 256, 0);
  gemm(o1, p_m[6], mbv + 256, nullptr, vb, 64, 256, 256, 0);
  attention_kernel<<<512, 256, 0, stream>>>(qb, kb, vb, Ob, 16, 16);
  gemm(Ob, p_m[7], mbo + 256, Ob, o2, 64, 256, 256, 2);

  // ---- MAB 3: self ----
  gemm(o2, p_m[8], mbq + 512, nullptr, qb, 64, 256, 256, 0);
  gemm(o2, p_m[9], mbk + 512, nullptr, kb, 64, 256, 256, 0);
  gemm(o2, p_m[10], mbv + 512, nullptr, vb, 64, 256, 256, 0);
  attention_kernel<<<512, 256, 0, stream>>>(qb, kb, vb, Ob, 16, 16);
  gemm(Ob, p_m[11], mbo + 512, Ob, o3, 64, 256, 256, 2);

  // ---- final projection -> d_out (4,16,12,32) ----
  gemm(o3, p_finW, finb, nullptr, (float*)d_out, 64, 384, 256, 0);
}
